// OutlookAtt_24094766530795
// MI455X (gfx1250) — matrix operand order repaired and checked by reading, not compiled, not run on hardware
//
#include <hip/hip_runtime.h>
#include <hip/hip_bf16.h>
#include <stdint.h>

// ---------------- problem constants ----------------
#define IMG_N   16
#define C_DIM   192
#define HEADS   6
#define CH      32
#define HDIM    56
#define HW      3136          // 56*56
#define NA      486           // 81*6 attention logits per position
#define NAP     496           // NA padded to 16
#define KK      9
#define ROWS    200           // padded LDS row stride in halves (192 + 8 pad)
#define SCALE   0.17677669529663687f  // 32^-0.5

typedef _Float16 h16v __attribute__((ext_vector_type(16)));
typedef _Float16 h8v  __attribute__((ext_vector_type(8)));
typedef float    f8v  __attribute__((ext_vector_type(8)));
typedef float    f4v  __attribute__((ext_vector_type(4)));
typedef uint32_t u4v  __attribute__((ext_vector_type(4)));

// Build a 16-half WMMA fragment from two 8-half (128-bit) chunks.
__device__ __forceinline__ h16v load_frag(const _Float16* p, int lo, int hi) {
    h8v a = *(const h8v*)(p + lo);
    h8v b = *(const h8v*)(p + hi);
    return __builtin_shufflevector(a, b, 0,1,2,3,4,5,6,7,8,9,10,11,12,13,14,15);
}

__device__ __forceinline__ f8v wmma_f16(h16v a, h16v b, f8v c) {
    return __builtin_amdgcn_wmma_f32_16x16x32_f16(false, a, false, b, (short)0, c,
                                                  false, false);
}

// ============================================================================
// Kernel 0: one-time weight prep.  f32 [k][n] -> f16 transposed [n][k].
// Wa padded with zeros to 496 columns. Total ~340 KB, cost negligible.
// ============================================================================
__global__ __launch_bounds__(256) void k0_prep(
    const float* __restrict__ Wv, const float* __restrict__ Wa,
    const float* __restrict__ Wp,
    _Float16* __restrict__ Wv16t, _Float16* __restrict__ Wa16t,
    _Float16* __restrict__ Wp16t)
{
    const int NV = C_DIM * C_DIM;            // 36864
    const int NW = NAP * C_DIM;              // 95232
    int i = blockIdx.x * 256 + threadIdx.x;  // total 168960
    if (i < NV) {
        int n = i / C_DIM, c = i - n * C_DIM;
        Wv16t[i] = (_Float16)Wv[c * C_DIM + n];
    } else if (i < NV + NW) {
        int j = i - NV;
        int n = j / C_DIM, c = j - n * C_DIM;
        Wa16t[j] = (n < NA) ? (_Float16)Wa[c * NA + n] : (_Float16)0.0f;
    } else if (i < NV + NW + NV) {
        int j = i - NV - NW;
        int n = j / C_DIM, c = j - n * C_DIM;
        Wp16t[j] = (_Float16)Wp[c * C_DIM + n];
    }
}

// ============================================================================
// Kernel 1: fused dual GEMM.  A = x (f32, K-major) staged->f16 LDS once;
// B fragments loaded straight from pre-transposed f16 weights (L2-resident).
//   v16[m,192]  = x @ Wv            (f16 out)
//   attL[m,486] = x @ Wa + ba       (f32 out, softmaxed later in place)
// Block: 128 threads = 4 waves; 64 m-rows per block (16 per wave); 43 n-tiles.
// ============================================================================
__global__ __launch_bounds__(128) void k1_dualgemm(
    const float* __restrict__ x,
    const _Float16* __restrict__ Wv16t, const _Float16* __restrict__ Wa16t,
    const float* __restrict__ ba,
    _Float16* __restrict__ v16, float* __restrict__ attL)
{
    __shared__ _Float16 As[64 * ROWS];       // 25.6 KB (only LDS use)

    const int tid  = threadIdx.x;
    const int wid  = tid >> 5;
    const int lane = tid & 31;
    const int col  = lane & 15;
    const int sel  = lane >> 4;              // 0: lanes 0-15, 1: lanes 16-31

    const int  blk   = blockIdx.x;           // 784 blocks
    const int  nimg  = blk / 49;
    const int  hwb   = (blk % 49) * 64;
    const long mbase = (long)nimg * HW + hwb;

    // Stage A tile: 64 rows (positions) x 192 ch. float4 along hw (coalesced
    // 16B per lane), convert, scatter 4 f16 rows into LDS.
    for (int i = tid; i < 64 * 192 / 4; i += 128) {
        int r = (i & 15) * 4, c = i >> 4;
        f4v v = *(const f4v*)(x + ((long)nimg * C_DIM + c) * HW + hwb + r);
        As[(r + 0) * ROWS + c] = (_Float16)v[0];
        As[(r + 1) * ROWS + c] = (_Float16)v[1];
        As[(r + 2) * ROWS + c] = (_Float16)v[2];
        As[(r + 3) * ROWS + c] = (_Float16)v[3];
    }
    __syncthreads();

    const int arow = wid * 16 + col;

    for (int nt = 0; nt < 43; ++nt) {
        int ncol;
        const _Float16* bcol;
        if (nt < 12) {
            ncol = nt * 16 + col;
            bcol = Wv16t + ncol * C_DIM;
        } else {
            ncol = (nt - 12) * 16 + col;
            bcol = Wa16t + ncol * C_DIM;     // rows up to 495 exist (zero tail)
        }
        // Prefetch next tile's B column while this tile computes.
        if (nt + 1 < 43) {
            const _Float16* nb = (nt + 1 < 12)
                ? Wv16t + ((nt + 1) * 16 + col) * C_DIM
                : Wa16t + ((nt + 1 - 12) * 16 + col) * C_DIM;
            __builtin_prefetch(nb, 0, 3);
        }

        f8v acc = {};
        #pragma unroll
        for (int kt = 0; kt < 6; ++kt) {     // K = 192 = 6 x 32
            // A 16x32 f16: lanes 0-15 hold K0-7,K16-23; lanes 16-31 K8-15,K24-31
            int al = arow * ROWS + kt * 32 + sel * 8;
            h16v a = load_frag(As, al, al + 16);
            // B 32x16 f16: lanes 0-15 hold K0-7,K16-23; lanes 16-31 K8-15,K24-31
            int bl = kt * 32 + sel * 8;
            h16v b = load_frag(bcol, bl, bl + 16);  // global_load_b128 x2 (L2 hit)
            acc = wmma_f16(a, b, acc);
        }

        // C layout: lane holds column n = lane&15; VGPR r holds row m = r + 8*sel
        if (nt < 12) {
            #pragma unroll
            for (int r = 0; r < 8; ++r) {
                long m = mbase + wid * 16 + r + 8 * sel;
                v16[m * C_DIM + ncol] = (_Float16)acc[r];
            }
        } else if (ncol < NA) {
            float bb = ba[ncol];
            #pragma unroll
            for (int r = 0; r < 8; ++r) {
                long m = mbase + wid * 16 + r + 8 * sel;
                attL[m * (long)NA + ncol] = acc[r] + bb;
            }
        }
    }
}

// ============================================================================
// Kernel 2: in-place softmax over rows of 9 (scale applied pre-softmax).
// ============================================================================
__global__ __launch_bounds__(256) void k2_softmax(float* __restrict__ attL, int R)
{
    int row = blockIdx.x * 256 + threadIdx.x;
    if (row >= R) return;
    float* p = attL + (long)row * 9;
    float v[9];
    float mx = -1e30f;
    #pragma unroll
    for (int i = 0; i < 9; ++i) { v[i] = p[i]; mx = fmaxf(mx, v[i]); }
    float s = 0.0f;
    #pragma unroll
    for (int i = 0; i < 9; ++i) { v[i] = __expf((v[i] - mx) * SCALE); s += v[i]; }
    float inv = 1.0f / s;
    #pragma unroll
    for (int i = 0; i < 9; ++i) p[i] = v[i] * inv;
}

// ============================================================================
// Kernel 3: attention apply + fold, in gather form.
//   folded[q,c] = sum_{k,l} att[q-off_k, head, k, l] * v[q - off_k + off_l, c]
//               = sum_{d in 5x5} w_d * v[q+d, c],  w_d = sum_{l-k=d} att[...]
// Block: 8x8 spatial tile, v halo tile (12x12x192 f16) staged in LDS.
// ============================================================================
__global__ __launch_bounds__(256) void k3_apply_fold(
    const float* __restrict__ attL, const _Float16* __restrict__ v16,
    _Float16* __restrict__ folded)
{
    __shared__ _Float16 Vs[144 * ROWS];      // 57.6 KB

    const int tid  = threadIdx.x;
    const int nimg = blockIdx.z;
    const int hb   = blockIdx.y * 8;
    const int wb   = blockIdx.x * 8;

    // Fill v halo tile (zero outside image). 144 rows x 24 16B-chunks, both
    // sides 16B aligned (row strides 384B global / 400B LDS).
    uint32_t* vs32 = (uint32_t*)Vs;
    const uint32_t* v32 = (const uint32_t*)v16;
    for (int i = tid; i < 144 * 24; i += 256) {
        int rr = i / 24, u = (i - rr * 24) * 4;
        int ih = rr / 12, iw = rr - ih * 12;
        int hh = hb + ih - 2, ww = wb + iw - 2;
        u4v val = {};
        if ((unsigned)hh < (unsigned)HDIM && (unsigned)ww < (unsigned)HDIM)
            val = *(const u4v*)(v32 + ((long)nimg * HW + hh * HDIM + ww) * 96 + u);
        *(u4v*)(vs32 + rr * 100 + u) = val;
    }
    __syncthreads();

    for (int u = tid; u < 64 * HEADS; u += 256) {
        int q = u & 63, head = u >> 6;
        int lh = q >> 3, lw = q & 7;
        int h = hb + lh, w = wb + lw;

        float wsum[25];
        #pragma unroll
        for (int d = 0; d < 25; ++d) wsum[d] = 0.0f;

        #pragma unroll
        for (int k = 0; k < 9; ++k) {
            int ki = k / 3, kj = k - ki * 3;
            int ph = h - (ki - 1), pw = w - (kj - 1);
            if ((unsigned)ph < (unsigned)HDIM && (unsigned)pw < (unsigned)HDIM) {
                const float* ar = attL +
                    (((long)nimg * HW + ph * HDIM + pw) * NA + head * 81 + k * 9);
                #pragma unroll
                for (int l = 0; l < 9; ++l) {
                    int li = l / 3, lj = l - li * 3;
                    wsum[(li - ki + 2) * 5 + (lj - kj + 2)] += ar[l];
                }
            }
        }

        float acc[CH];
        #pragma unroll
        for (int c = 0; c < CH; ++c) acc[c] = 0.0f;

        #pragma unroll
        for (int d = 0; d < 25; ++d) {
            float wd = wsum[d];
            int di = d / 5, dj = d - di * 5;
            int rr = (lh + di) * 12 + (lw + dj);
            const uint32_t* vp = (const uint32_t*)&Vs[rr * ROWS + head * CH];
            #pragma unroll
            for (int ci = 0; ci < CH / 2; ++ci) {
                uint32_t pk = vp[ci];
                _Float16 lo = ((const _Float16*)&pk)[0];
                _Float16 hi = ((const _Float16*)&pk)[1];
                acc[2 * ci]     += wd * (float)lo;
                acc[2 * ci + 1] += wd * (float)hi;
            }
        }

        h8v* op = (h8v*)(folded + (((long)nimg * HW + h * HDIM + w) * C_DIM + head * CH));
        #pragma unroll
        for (int g = 0; g < 4; ++g) {
            h8v o;
            #pragma unroll
            for (int e = 0; e < 8; ++e) o[e] = (_Float16)acc[g * 8 + e];
            op[g] = o;                        // global_store_b128 x4
        }
    }
}

// ============================================================================
// Kernel 4: projection GEMM  out = folded @ Wp + bp  (NCHW f32 output).
// A (f16 row-major) and B (pre-transposed f16) both loaded straight from
// global with 128-bit loads; no LDS, no barriers.
// ============================================================================
__global__ __launch_bounds__(128) void k4_proj(
    const _Float16* __restrict__ folded, const _Float16* __restrict__ Wp16t,
    const float* __restrict__ bp, float* __restrict__ out)
{
    const int tid  = threadIdx.x;
    const int wid  = tid >> 5;
    const int lane = tid & 31;
    const int col  = lane & 15;
    const int sel  = lane >> 4;

    const int  blk   = blockIdx.x;           // 784 blocks
    const int  nimg  = blk / 49;
    const int  hwb   = (blk % 49) * 64;
    const long mbase = (long)nimg * HW + hwb;

    const _Float16* aptr = folded + (mbase + wid * 16 + col) * (long)C_DIM;

    for (int nt = 0; nt < 12; ++nt) {
        int n = nt * 16 + col;
        const _Float16* bcol = Wp16t + n * C_DIM;

        f8v acc = {};
        #pragma unroll
        for (int kt = 0; kt < 6; ++kt) {
            int al = kt * 32 + sel * 8;
            h16v a = load_frag(aptr, al, al + 16);     // global_load_b128 x2
            int bl = kt * 32 + sel * 8;
            h16v b = load_frag(bcol, bl, bl + 16);     // global_load_b128 x2
            acc = wmma_f16(a, b, acc);
        }

        float bb = bp[n];
        long base = ((long)nimg * C_DIM + n) * HW + hwb + wid * 16 + 8 * sel;
        f4v o0 = { acc[0] + bb, acc[1] + bb, acc[2] + bb, acc[3] + bb };
        f4v o1 = { acc[4] + bb, acc[5] + bb, acc[6] + bb, acc[7] + bb };
        *(f4v*)(out + base)     = o0;
        *(f4v*)(out + base + 4) = o1;
    }
}

// ============================================================================
extern "C" void kernel_launch(void* const* d_in, const int* in_sizes, int n_in,
                              void* d_out, int out_size, void* d_ws, size_t ws_size,
                              hipStream_t stream)
{
    const float* x  = (const float*)d_in[0];
    const float* Wv = (const float*)d_in[1];
    const float* Wa = (const float*)d_in[2];
    const float* ba = (const float*)d_in[3];
    const float* Wp = (const float*)d_in[4];
    const float* bp = (const float*)d_in[5];
    float* out = (float*)d_out;

    char* ws = (char*)d_ws;
    const size_t V16_BYTES  = (size_t)IMG_N * HW * C_DIM * 2;   // 19,267,584
    const size_t ATT_BYTES  = (size_t)IMG_N * HW * NA * 4;      // 97,542,144
    const size_t FLD_BYTES  = V16_BYTES;                        // 19,267,584
    const size_t WV_BYTES   = (size_t)C_DIM * C_DIM * 2;        // 73,728
    const size_t WA_BYTES   = (size_t)NAP * C_DIM * 2;          // 190,464

    _Float16* v16    = (_Float16*)(ws);
    float*    attL   = (float*)(ws + V16_BYTES);
    _Float16* folded = (_Float16*)(ws + V16_BYTES + ATT_BYTES);
    _Float16* Wv16t  = (_Float16*)(ws + V16_BYTES + ATT_BYTES + FLD_BYTES);
    _Float16* Wa16t  = (_Float16*)(ws + V16_BYTES + ATT_BYTES + FLD_BYTES + WV_BYTES);
    _Float16* Wp16t  = (_Float16*)(ws + V16_BYTES + ATT_BYTES + FLD_BYTES + WV_BYTES + WA_BYTES);
    // total workspace: ~136.4 MB

    // Weight prep: (36864 + 95232 + 36864) elements
    k0_prep<<<(168960 + 255) / 256, 256, 0, stream>>>(Wv, Wa, Wp, Wv16t, Wa16t, Wp16t);

    // GEMM blocks: 50176 rows / 64 = 784
    k1_dualgemm<<<784, 128, 0, stream>>>(x, Wv16t, Wa16t, ba, v16, attL);

    const int R = IMG_N * HW * HEADS * KK;   // 2,709,504 softmax rows of 9
    k2_softmax<<<(R + 255) / 256, 256, 0, stream>>>(attL, R);

    dim3 g3(7, 7, IMG_N);                    // 8x8 spatial tiles
    k3_apply_fold<<<g3, 256, 0, stream>>>(attL, v16, folded);

    k4_proj<<<784, 128, 0, stream>>>(folded, Wp16t, bp, out);
}
